// ConvNet_41283225649624
// MI455X (gfx1250) — compile-verified, matrix operand order checked by reading, not run on hardware
//
#include <hip/hip_runtime.h>
#include <hip/hip_bf16.h>

typedef __attribute__((ext_vector_type(16))) _Float16 v16h;
typedef __attribute__((ext_vector_type(8)))  float    v8f;

#define N_NODES  8192
#define N_GRAPHS 16
#define NPG      512      // nodes per graph
#define KNN      80
#define F0       6
#define F0P      32       // F0 zero-padded to one WMMA k-block
#define HID      128
#define HID6     768
#define NLAB     3
#define LEAK     0.01f

// ---------------------------------------------------------------------------
// Fragment packing helpers.
// ISA 16-bit A/B layout per lane (16 halves):
//   hi=0 (lanes 0-15):  K = {0..7, 16..23}
//   hi=1 (lanes 16-31): K = {8..15, 24..31}
// For kk = k % 32:  hi = (kk>>3)&1,  pos = (kk&7) + (kk>=16 ? 8 : 0)
// Packed element address = base + row*K + kb*32 + hi*16 + pos   (halves)
// so each lane's fragment is one contiguous 32-byte chunk.
// ---------------------------------------------------------------------------
__device__ __forceinline__ int frag_off(int kk) {
    const int hi  = (kk >> 3) & 1;
    const int pos = (kk & 7) + ((kk >= 16) ? 8 : 0);
    return hi * 16 + pos;
}

// ---------------------------------------------------------------------------
// KNN: one block (512 threads) per node; distances in LDS, 80 argmin extracts.
// norm = dinv[src]*dinv[tgt] = 1/80 exactly (every in-degree == 80).
// ---------------------------------------------------------------------------
__global__ void knn_kernel(const float* __restrict__ x, int* __restrict__ nbrs) {
    __shared__ float sd[NPG];
    __shared__ float rd[NPG];
    __shared__ int   ri[NPG];
    const int node = blockIdx.x;
    const int base = (node >> 9) << 9;      // graph start
    const int t    = threadIdx.x;           // 0..511
    const int j    = base + t;

    const float p0 = x[node * F0 + 0], p1 = x[node * F0 + 1];
    const float p2 = x[node * F0 + 2], p3 = x[node * F0 + 3];
    float d0 = x[j * F0 + 0] - p0;
    float d1 = x[j * F0 + 1] - p1;
    float d2 = x[j * F0 + 2] - p2;
    float d3 = x[j * F0 + 3] - p3;
    float d  = d0 * d0 + d1 * d1 + d2 * d2 + d3 * d3;
    if (j == node) d = 3.4e38f;
    sd[t] = d;
    __syncthreads();

    for (int sel = 0; sel < KNN; ++sel) {
        rd[t] = sd[t];
        ri[t] = t;
        __syncthreads();
        for (int s = NPG / 2; s > 0; s >>= 1) {
            if (t < s) {
                if (rd[t + s] < rd[t]) { rd[t] = rd[t + s]; ri[t] = ri[t + s]; }
            }
            __syncthreads();
        }
        const int w = ri[0];
        if (t == 0) nbrs[node * KNN + sel] = base + w;
        if (t == w) sd[t] = 3.4e38f;
        __syncthreads();
    }
}

// ---------------------------------------------------------------------------
// Mean-over-neighbors aggregation: out[i,f] = (1/80) * sum_e h[nbr[i,e], f]
// ---------------------------------------------------------------------------
__global__ void aggregate_kernel(const float* __restrict__ h,
                                 const int* __restrict__ nbrs,
                                 float* __restrict__ out, int F) {
    const int node = blockIdx.x;
    const int f    = threadIdx.x;
    if (f >= F) return;
    const int* nb = nbrs + node * KNN;
    float acc = 0.f;
#pragma unroll 8
    for (int e = 0; e < KNN; ++e)
        acc += h[(size_t)nb[e] * F + f];
    out[(size_t)node * F + f] = acc * (1.0f / (float)KNN);
}

// ---------------------------------------------------------------------------
// Pack a row-major f32 activation matrix (rows x Ksrc) into fragment-ordered
// f16 (rows x K, K multiple of 32, zero padded).  block = K threads.
// ---------------------------------------------------------------------------
__global__ void pack_act_kernel(const float* __restrict__ src,
                                _Float16* __restrict__ dst,
                                int K, int Ksrc) {
    const int row = blockIdx.x;
    const int k   = threadIdx.x;          // 0..K-1
    const int kb  = k >> 5;
    const float v = (k < Ksrc) ? src[(size_t)row * Ksrc + k] : 0.f;
    dst[(size_t)row * K + kb * 32 + frag_off(k & 31)] = (_Float16)v;
}

// ---------------------------------------------------------------------------
// Pack weights W (nterms x Ksrc x N, f32 row-major) into fragment-ordered f16
// B operands, column-major-by-tile:  dst[(t*N + n)*K + kb*32 + frag_off]
// ---------------------------------------------------------------------------
__global__ void pack_w_kernel(const float* __restrict__ w,
                              _Float16* __restrict__ dst,
                              int K, int Ksrc, int N, int nterms) {
    const int idx = blockIdx.x * blockDim.x + threadIdx.x;
    const int total = nterms * K * N;
    if (idx >= total) return;
    const int t   = idx / (K * N);
    const int rem = idx - t * K * N;
    const int k   = rem / N;
    const int n   = rem - k * N;
    const float v = (k < Ksrc) ? w[((size_t)t * Ksrc + k) * N + n] : 0.f;
    const int kb  = k >> 5;
    dst[((size_t)t * N + n) * K + kb * 32 + frag_off(k & 31)] = (_Float16)v;
}

// ---------------------------------------------------------------------------
// WMMA GEMM:  Y = leaky( sum_{t<NT} A_t @ B_t + bias )
//   Operands pre-packed f16 fragments; f32 accumulation; K compile-time.
//   Each wave owns a 16x16 tile: per k-step = 2 x 32B vector loads + 1 wmma.
//   block = 256 (8 waves over 128 columns); grid(M/16, N/128)
// ---------------------------------------------------------------------------
template <int K, int NT>
__global__ void tag_gemm_kernel(const _Float16* __restrict__ PA0,
                                const _Float16* __restrict__ PA1,
                                const _Float16* __restrict__ PA2,
                                const _Float16* __restrict__ PB,
                                const float* __restrict__ bias,
                                float* __restrict__ Y, int N) {
    const int wave = threadIdx.x >> 5;
    const int lane = threadIdx.x & 31;
    const int hi   = lane >> 4;
    const int ln   = lane & 15;
    const int row0 = blockIdx.x * 16;
    const int col  = blockIdx.y * 128 + wave * 16 + ln;

    const _Float16* PAs[3] = { PA0, PA1, PA2 };

    v8f c;
#pragma unroll
    for (int i = 0; i < 8; ++i) c[i] = 0.f;

#pragma unroll
    for (int t = 0; t < NT; ++t) {
        const _Float16* pa_row = PAs[t] + (size_t)(row0 + ln) * K + hi * 16;
        const _Float16* pb_col = PB + ((size_t)t * N + col) * K + hi * 16;
#pragma unroll
        for (int kb = 0; kb < K / 32; ++kb) {
            const v16h a = *reinterpret_cast<const v16h*>(pa_row + kb * 32);
            const v16h b = *reinterpret_cast<const v16h*>(pb_col + kb * 32);
            c = __builtin_amdgcn_wmma_f32_16x16x32_f16(
                    false, a, false, b, (short)0, c, false, false);
        }
    }

    // C/D layout: VGPR r -> M = r + 8*hi, N = lane&15
    const float bv = bias[col];
#pragma unroll
    for (int r = 0; r < 8; ++r) {
        const int row = row0 + r + 8 * hi;
        float y = c[r] + bv;
        y = (y > 0.f) ? y : LEAK * y;
        Y[(size_t)row * N + col] = y;
    }
}

// ---------------------------------------------------------------------------
// Per-graph mean+max pooling into g columns [coloff..+127]=mean, [+128..]=max
// ---------------------------------------------------------------------------
__global__ void pool_kernel(const float* __restrict__ h, float* __restrict__ g,
                            int coloff) {
    const int gr = blockIdx.x;
    const int f  = threadIdx.x;          // 0..127
    const float* p = h + (size_t)gr * NPG * HID + f;
    float s = 0.f, m = -3.4e38f;
    for (int n = 0; n < NPG; ++n) {
        const float v = p[(size_t)n * HID];
        s += v;
        m = fmaxf(m, v);
    }
    g[gr * HID6 + coloff + f]       = s * (1.0f / (float)NPG);
    g[gr * HID6 + coloff + HID + f] = m;
}

// ---------------------------------------------------------------------------
// BatchNorm (inference) in place on g : (16 x 768)
// ---------------------------------------------------------------------------
__global__ void bn_kernel(float* __restrict__ g,
                          const float* __restrict__ gamma,
                          const float* __restrict__ beta,
                          const float* __restrict__ mean,
                          const float* __restrict__ var) {
    const int idx = blockIdx.x * blockDim.x + threadIdx.x;
    if (idx >= N_GRAPHS * HID6) return;
    const int c = idx % HID6;
    const float v = g[idx];
    g[idx] = (v - mean[c]) * rsqrtf(var[c] + 1e-5f) * gamma[c] + beta[c];
}

// ---------------------------------------------------------------------------
// Output head: (16 x 768) @ (768 x 3) + b
// ---------------------------------------------------------------------------
__global__ void head_out_kernel(const float* __restrict__ g,
                                const float* __restrict__ ow,
                                const float* __restrict__ ob,
                                float* __restrict__ out) {
    const int t = threadIdx.x;
    if (t >= N_GRAPHS * NLAB) return;
    const int i = t / NLAB, j = t % NLAB;
    float acc = ob[j];
    for (int k = 0; k < HID6; ++k)
        acc += g[i * HID6 + k] * ow[k * NLAB + j];
    out[i * NLAB + j] = acc;
}

// ---------------------------------------------------------------------------
extern "C" void kernel_launch(void* const* d_in, const int* in_sizes, int n_in,
                              void* d_out, int out_size, void* d_ws, size_t ws_size,
                              hipStream_t stream) {
    const float* x       = (const float*)d_in[0];
    // d_in[1] = batch (contiguous; layout is hardcoded)
    const float* conv1_w = (const float*)d_in[2];
    const float* conv1_b = (const float*)d_in[3];
    const float* conv2_w = (const float*)d_in[4];
    const float* conv2_b = (const float*)d_in[5];
    const float* conv3_w = (const float*)d_in[6];
    const float* conv3_b = (const float*)d_in[7];
    const float* bn_g    = (const float*)d_in[8];
    const float* bn_b    = (const float*)d_in[9];
    const float* bn_m    = (const float*)d_in[10];
    const float* bn_v    = (const float*)d_in[11];
    const float* lin_w   = (const float*)d_in[12];
    const float* lin_b   = (const float*)d_in[13];
    const float* out_w   = (const float*)d_in[14];
    const float* out_b   = (const float*)d_in[15];
    float* out = (float*)d_out;

    // workspace carve-up (256B aligned)
    char* ws = (char*)d_ws;
    size_t off = 0;
    auto alloc = [&](size_t bytes) {
        void* p = ws + off;
        off = (off + bytes + 255) & ~(size_t)255;
        return p;
    };
    int*      nbrs = (int*)     alloc((size_t)N_NODES * KNN * sizeof(int));
    float*    a1   = (float*)   alloc((size_t)N_NODES * HID * sizeof(float));
    float*    a2   = (float*)   alloc((size_t)N_NODES * HID * sizeof(float));
    float*    hA   = (float*)   alloc((size_t)N_NODES * HID * sizeof(float));
    float*    hB   = (float*)   alloc((size_t)N_NODES * HID * sizeof(float));
    float*    g1   = (float*)   alloc((size_t)N_GRAPHS * HID6 * sizeof(float));
    float*    g2   = (float*)   alloc((size_t)N_GRAPHS * HID6 * sizeof(float));
    _Float16* PX   = (_Float16*)alloc((size_t)N_NODES * F0P  * sizeof(_Float16));
    _Float16* PA1  = (_Float16*)alloc((size_t)N_NODES * HID  * sizeof(_Float16));
    _Float16* PA2  = (_Float16*)alloc((size_t)N_NODES * HID  * sizeof(_Float16));
    _Float16* PH   = (_Float16*)alloc((size_t)N_NODES * HID  * sizeof(_Float16));
    _Float16* PW1  = (_Float16*)alloc((size_t)3 * F0P * HID  * sizeof(_Float16));
    _Float16* PW2  = (_Float16*)alloc((size_t)3 * HID * HID  * sizeof(_Float16));
    _Float16* PW3  = (_Float16*)alloc((size_t)3 * HID * HID  * sizeof(_Float16));
    _Float16* PLW  = (_Float16*)alloc((size_t)5 * HID6 * HID6 * sizeof(_Float16));
    _Float16* Pg   = (_Float16*)alloc((size_t)N_GRAPHS * HID6 * sizeof(_Float16));

    // 0) pack all weights into fragment-ordered f16 (conv1 K padded 6 -> 32)
    pack_w_kernel<<<(3 * F0P * HID + 255) / 256, 256, 0, stream>>>(
        conv1_w, PW1, F0P, F0, HID, 3);
    pack_w_kernel<<<(3 * HID * HID + 255) / 256, 256, 0, stream>>>(
        conv2_w, PW2, HID, HID, HID, 3);
    pack_w_kernel<<<(3 * HID * HID + 255) / 256, 256, 0, stream>>>(
        conv3_w, PW3, HID, HID, HID, 3);
    pack_w_kernel<<<(5 * HID6 * HID6 + 255) / 256, 256, 0, stream>>>(
        lin_w, PLW, HID6, HID6, HID6, 5);

    // 1) KNN graph
    knn_kernel<<<N_NODES, NPG, 0, stream>>>(x, nbrs);

    // 2) conv1: K = 6 (padded to 32)
    aggregate_kernel<<<N_NODES, HID, 0, stream>>>(x,  nbrs, a1, F0);
    aggregate_kernel<<<N_NODES, HID, 0, stream>>>(a1, nbrs, a2, F0);
    pack_act_kernel<<<N_NODES, F0P, 0, stream>>>(x,  PX,  F0P, F0);
    pack_act_kernel<<<N_NODES, F0P, 0, stream>>>(a1, PA1, F0P, F0);
    pack_act_kernel<<<N_NODES, F0P, 0, stream>>>(a2, PA2, F0P, F0);
    tag_gemm_kernel<F0P, 3><<<dim3(N_NODES / 16, 1), 256, 0, stream>>>(
        PX, PA1, PA2, PW1, conv1_b, hA, HID);
    pool_kernel<<<N_GRAPHS, HID, 0, stream>>>(hA, g1, 0);

    // 3) conv2: K = 128
    aggregate_kernel<<<N_NODES, HID, 0, stream>>>(hA, nbrs, a1, HID);
    aggregate_kernel<<<N_NODES, HID, 0, stream>>>(a1, nbrs, a2, HID);
    pack_act_kernel<<<N_NODES, HID, 0, stream>>>(hA, PH,  HID, HID);
    pack_act_kernel<<<N_NODES, HID, 0, stream>>>(a1, PA1, HID, HID);
    pack_act_kernel<<<N_NODES, HID, 0, stream>>>(a2, PA2, HID, HID);
    tag_gemm_kernel<HID, 3><<<dim3(N_NODES / 16, 1), 256, 0, stream>>>(
        PH, PA1, PA2, PW2, conv2_b, hB, HID);
    pool_kernel<<<N_GRAPHS, HID, 0, stream>>>(hB, g1, 2 * HID);

    // 4) conv3: K = 128
    aggregate_kernel<<<N_NODES, HID, 0, stream>>>(hB, nbrs, a1, HID);
    aggregate_kernel<<<N_NODES, HID, 0, stream>>>(a1, nbrs, a2, HID);
    pack_act_kernel<<<N_NODES, HID, 0, stream>>>(hB, PH,  HID, HID);
    pack_act_kernel<<<N_NODES, HID, 0, stream>>>(a1, PA1, HID, HID);
    pack_act_kernel<<<N_NODES, HID, 0, stream>>>(a2, PA2, HID, HID);
    tag_gemm_kernel<HID, 3><<<dim3(N_NODES / 16, 1), 256, 0, stream>>>(
        PH, PA1, PA2, PW3, conv3_b, hA, HID);
    pool_kernel<<<N_GRAPHS, HID, 0, stream>>>(hA, g1, 4 * HID);

    // 5) BatchNorm on pooled features, then pack for MLP
    bn_kernel<<<(N_GRAPHS * HID6 + 255) / 256, 256, 0, stream>>>(
        g1, bn_g, bn_b, bn_m, bn_v);
    pack_act_kernel<<<N_GRAPHS, HID6, 0, stream>>>(g1, Pg, HID6, HID6);

    // 6) MLP head: 5 x (768 -> 768) leaky relu, WMMA (M = 16 -> one tile row)
    for (int i = 0; i < 5; ++i) {
        tag_gemm_kernel<HID6, 1><<<dim3(1, HID6 / 128), 256, 0, stream>>>(
            Pg, Pg, Pg, PLW + (size_t)i * HID6 * HID6, lin_b + (size_t)i * HID6,
            g2, HID6);
        pack_act_kernel<<<N_GRAPHS, HID6, 0, stream>>>(g2, Pg, HID6, HID6);
    }

    // 7) output projection 768 -> 3
    head_out_kernel<<<1, 64, 0, stream>>>(g2, out_w, out_b, out);
}
